// DeBERTa_26620207300905
// MI455X (gfx1250) — compile-verified, matrix operand order checked by reading, not run on hardware
//
#include <hip/hip_runtime.h>
#include <cmath>

// ---------------- problem constants ----------------
constexpr int B_ = 8, S_ = 512, HID_ = 768, L_ = 12, NH_ = 12, D_ = 64;
constexpr int FF_ = 3072, cM = 512, TWOM = 1024;
constexpr float SCALE_ = 0.07216878364870322f;   // 1/sqrt(3*64)

typedef __attribute__((ext_vector_type(16))) __bf16 v16bf;
typedef __attribute__((ext_vector_type(8)))  __bf16 v8bf;
typedef __attribute__((ext_vector_type(8)))  float  v8f;
typedef __attribute__((ext_vector_type(4)))  float  f32x4;

union FragU { v16bf v; v8bf h[2]; };

enum { EPI_F32 = 0, EPI_BF16, EPI_QKV, EPI_PK, EPI_CTX, EPI_RESID, EPI_GELU };

// =====================================================================
// Generic bf16 WMMA GEMM: C = A[M,K] * B + epilogue
//   AF32 : A source is fp32 (converted to bf16 while staging to LDS)
//   TRB  : B source stored [N,K] row-major (already N-major) - bf16
//   BF32 : (when !TRB) B source [K,N] is fp32 weights (else bf16)
// Block tile BM x BN x 32, 256 threads = 8 waves (4x2 wave grid),
// each wave computes (BM/4)x(BN/2) via 16x16x32 WMMA sub-tiles.
// Double-buffered LDS (one barrier per K-step) + L2 prefetch of K+2 tile.
// =====================================================================
template<int EPI, bool AF32, bool TRB, bool BF32, int BM, int BN>
__global__ __launch_bounds__(256)
void gemm_wmma(const void* __restrict__ Ap, const void* __restrict__ Bp,
               void* __restrict__ Cp, const float* __restrict__ bias,
               int N, int K, int ldA, int ldB,
               long long sA, long long sB, long long sC, int bmod)
{
    constexpr int BK  = 32;
    constexpr int SA  = BK + 8;            // padded LDS row stride (elems)
    constexpr int SMT = BM / 64;           // sub-tiles per wave (M)
    constexpr int SNT = BN / 32;           // sub-tiles per wave (N)

    __shared__ __bf16 As[2][BM * SA];
    __shared__ __bf16 Bs[2][BN * SA];

    const int tid  = threadIdx.x;
    const int wave = tid >> 5;
    const int lane = tid & 31;
    const int bm   = blockIdx.y * BM;
    const int bn   = blockIdx.x * BN;
    const long long z = blockIdx.z;

    const char* Abase = (const char*)Ap + z * sA * (AF32 ? 4 : 2);
    const long long zb = bmod ? (z % bmod) : z;
    const char* Bbase = (const char*)Bp + zb * sB * ((!TRB && BF32) ? 4 : 2);

    const int wm = (wave >> 1) * (BM / 4);
    const int wn = (wave & 1) * (BN / 2);

    v8f acc[SMT][SNT];
    #pragma unroll
    for (int i = 0; i < SMT; ++i)
        #pragma unroll
        for (int j = 0; j < SNT; ++j)
            #pragma unroll
            for (int e = 0; e < 8; ++e) acc[i][j][e] = 0.0f;

    const int lrow = lane & 15;
    const int lk   = (lane >> 4) * 8;      // K offset of this lane's runs

    // ------- tile staging (global -> LDS, with fp32->bf16 convert) -------
    auto stage = [&](int buf, int k0) {
        __bf16* Asb = As[buf];
        __bf16* Bsb = Bs[buf];
        // A tile, row-major [BM][BK]
        if constexpr (!AF32) {
            const __bf16* A = (const __bf16*)Abase;
            for (int base = tid * 8; base < BM * BK; base += 256 * 8) {
                int r = base / BK, c = base % BK;
                const __bf16* src = A + (long long)(bm + r) * ldA + k0 + c;
                if (k0 + 2 * BK < K) __builtin_prefetch(src + 2 * BK, 0, 3);
                *(v8bf*)&Asb[r * SA + c] = *(const v8bf*)src;
            }
        } else {
            const float* A = (const float*)Abase;
            for (int base = tid * 8; base < BM * BK; base += 256 * 8) {
                int r = base / BK, c = base % BK;
                const float* src = A + (long long)(bm + r) * ldA + k0 + c;
                if (k0 + 2 * BK < K) __builtin_prefetch(src + 2 * BK, 0, 3);
                f32x4 x0 = ((const f32x4*)src)[0], x1 = ((const f32x4*)src)[1];
                v8bf o;
                #pragma unroll
                for (int j = 0; j < 4; ++j) { o[j] = (__bf16)x0[j]; o[4 + j] = (__bf16)x1[j]; }
                *(v8bf*)&Asb[r * SA + c] = o;
            }
        }
        // B tile staged N-major [BN][BK]
        if constexpr (TRB) {                       // B already [N,K] bf16
            const __bf16* Bt = (const __bf16*)Bbase;
            for (int base = tid * 8; base < BN * BK; base += 256 * 8) {
                int n = base / BK, c = base % BK;
                const __bf16* src = Bt + (long long)(bn + n) * ldB + k0 + c;
                if (k0 + 2 * BK < K) __builtin_prefetch(src + 2 * BK, 0, 3);
                *(v8bf*)&Bsb[n * SA + c] = *(const v8bf*)src;
            }
        } else if constexpr (BF32) {               // B [K,N] fp32 weights
            const float* Bf = (const float*)Bbase;
            for (int base = tid * 4; base < BK * BN; base += 256 * 4) {
                int kk = base / BN, n = base % BN;
                const float* src = Bf + (long long)(k0 + kk) * ldB + bn + n;
                if (k0 + 2 * BK < K) __builtin_prefetch(src + (long long)2 * BK * ldB, 0, 3);
                f32x4 xv = *(const f32x4*)src;
                #pragma unroll
                for (int j = 0; j < 4; ++j) Bsb[(n + j) * SA + kk] = (__bf16)xv[j];
            }
        } else {                                   // B [K,N] bf16 (V matrix)
            const __bf16* Bb = (const __bf16*)Bbase;
            for (int base = tid * 8; base < BK * BN; base += 256 * 8) {
                int kk = base / BN, n = base % BN;
                const __bf16* src = Bb + (long long)(k0 + kk) * ldB + bn + n;
                if (k0 + 2 * BK < K) __builtin_prefetch(src + (long long)2 * BK * ldB, 0, 3);
                v8bf xv = *(const v8bf*)src;
                #pragma unroll
                for (int j = 0; j < 8; ++j) Bsb[(n + j) * SA + kk] = xv[j];
            }
        }
    };

    const int nsteps = K / BK;
    stage(0, 0);
    __syncthreads();

    for (int s = 0; s < nsteps; ++s) {
        const int cur = s & 1;
        if (s + 1 < nsteps) stage(cur ^ 1, (s + 1) * BK);   // prefetch next tile into LDS

        const __bf16* Asb = As[cur];
        const __bf16* Bsb = Bs[cur];
        FragU a[SMT], bf[SNT];
        #pragma unroll
        for (int i = 0; i < SMT; ++i) {
            int r = wm + i * 16 + lrow;
            a[i].h[0] = *(const v8bf*)&Asb[r * SA + lk];
            a[i].h[1] = *(const v8bf*)&Asb[r * SA + 16 + lk];
        }
        #pragma unroll
        for (int j = 0; j < SNT; ++j) {
            int n = wn + j * 16 + lrow;
            bf[j].h[0] = *(const v8bf*)&Bsb[n * SA + lk];
            bf[j].h[1] = *(const v8bf*)&Bsb[n * SA + 16 + lk];
        }
        #pragma unroll
        for (int i = 0; i < SMT; ++i)
            #pragma unroll
            for (int j = 0; j < SNT; ++j)
                acc[i][j] = __builtin_amdgcn_wmma_f32_16x16x32_bf16(
                    false, a[i].v, false, bf[j].v, (short)0, acc[i][j], false, false);
        __syncthreads();
    }

    // ---------------- epilogue ----------------
    const int nl = lane & 15;
    const int mb = (lane >> 4) * 8;
    int ctx_b = 0, ctx_h = 0;
    if constexpr (EPI == EPI_CTX) { ctx_b = (int)(z / NH_); ctx_h = (int)(z % NH_); }
    #pragma unroll
    for (int i = 0; i < SMT; ++i)
        #pragma unroll
        for (int j = 0; j < SNT; ++j)
            #pragma unroll
            for (int r = 0; r < 8; ++r) {
                int m = bm + wm + i * 16 + mb + r;     // local row (per batch)
                int n = bn + wn + j * 16 + nl;
                float val = acc[i][j][r];
                if constexpr (EPI == EPI_F32) {
                    float* C = (float*)Cp + z * sC;
                    C[(long long)m * N + n] = val;
                } else if constexpr (EPI == EPI_BF16) {
                    __bf16* C = (__bf16*)Cp + z * sC;
                    C[(long long)m * N + n] = (__bf16)val;
                } else if constexpr (EPI == EPI_QKV) {
                    // [B*S, HID] -> [B, NH, S, D] bf16 (+bias); S=512, D=64 pow2
                    __bf16* C = (__bf16*)Cp;
                    int b = m >> 9, s = m & 511, hh = n >> 6, dd = n & 63;
                    C[(((long long)b * NH_ + hh) * S_ + s) * D_ + dd] =
                        (__bf16)(val + bias[n]);
                } else if constexpr (EPI == EPI_PK) {
                    // [2M, HID] -> [NH, 2M, D] bf16 (no bias)
                    __bf16* C = (__bf16*)Cp;
                    int hh = n >> 6, dd = n & 63;
                    C[((long long)hh * TWOM + m) * D_ + dd] = (__bf16)val;
                } else if constexpr (EPI == EPI_CTX) {
                    // per-batch [S, D] -> [B, S, HID] fp32
                    float* C = (float*)Cp;
                    C[((long long)(ctx_b * S_ + m)) * HID_ + ctx_h * D_ + n] = val;
                } else if constexpr (EPI == EPI_RESID) {
                    float* C = (float*)Cp;
                    long long idx = (long long)m * N + n;
                    C[idx] += val + bias[n];
                } else if constexpr (EPI == EPI_GELU) {
                    __bf16* C = (__bf16*)Cp;
                    float t = val + bias[n];
                    C[(long long)m * N + n] =
                        (__bf16)(0.5f * t * (1.0f + erff(t * 0.70710678118654752f)));
                }
            }
}

// =====================================================================
// block reductions (256 threads = 8 wave32)
// =====================================================================
__device__ __forceinline__ float blk_sum(float v, float* sh) {
    #pragma unroll
    for (int o = 16; o > 0; o >>= 1) v += __shfl_xor(v, o, 32);
    int w = threadIdx.x >> 5, l = threadIdx.x & 31;
    if (l == 0) sh[w] = v;
    __syncthreads();
    if (w == 0) {
        float r = (l < 8) ? sh[l] : 0.0f;
        #pragma unroll
        for (int o = 4; o > 0; o >>= 1) r += __shfl_xor(r, o, 32);
        if (l == 0) sh[0] = r;
    }
    __syncthreads();
    float res = sh[0];
    __syncthreads();
    return res;
}

__device__ __forceinline__ float blk_max(float v, float* sh) {
    #pragma unroll
    for (int o = 16; o > 0; o >>= 1) v = fmaxf(v, __shfl_xor(v, o, 32));
    int w = threadIdx.x >> 5, l = threadIdx.x & 31;
    if (l == 0) sh[w] = v;
    __syncthreads();
    if (w == 0) {
        float r = (l < 8) ? sh[l] : -3.4e38f;
        #pragma unroll
        for (int o = 4; o > 0; o >>= 1) r = fmaxf(r, __shfl_xor(r, o, 32));
        if (l == 0) sh[0] = r;
    }
    __syncthreads();
    float res = sh[0];
    __syncthreads();
    return res;
}

// =====================================================================
// embedding: x = tok_emb[id] + sinusoidal_pe + seg_emb[seg]
// =====================================================================
__global__ __launch_bounds__(256)
void embed_kernel(const int* __restrict__ tok, const int* __restrict__ seg,
                  const float* __restrict__ tok_emb, const float* __restrict__ seg_emb,
                  float* __restrict__ x)
{
    long long row = blockIdx.x;              // b*S + s
    int s = (int)(row % S_);
    int t = tok[row], g = seg[row];
    for (int c = threadIdx.x; c < HID_; c += 256) {
        int i = c >> 1;
        float freq = powf(10000.0f, -(2.0f * (float)i) / (float)HID_);
        float ang  = (float)s * freq;
        float pe   = (c & 1) ? cosf(ang) : sinf(ang);
        x[row * HID_ + c] = tok_emb[(long long)t * HID_ + c] + pe
                          + seg_emb[(long long)g * HID_ + c];
    }
}

// =====================================================================
// LayerNorm over HID=768 (one block per row), bf16 or fp32 output
// =====================================================================
template<bool OBF>
__global__ __launch_bounds__(256)
void ln_kernel(const float* __restrict__ x, const float* __restrict__ gm,
               const float* __restrict__ bt, void* __restrict__ outp)
{
    __shared__ float sh[8];
    long long row = blockIdx.x;
    const float* xr = x + row * HID_;
    float vals[3], s = 0.f, ss = 0.f;
    #pragma unroll
    for (int t = 0; t < 3; ++t) {
        float v = xr[threadIdx.x + t * 256];
        vals[t] = v; s += v; ss += v * v;
    }
    float mean = blk_sum(s, sh) * (1.0f / HID_);
    float var  = blk_sum(ss, sh) * (1.0f / HID_) - mean * mean;
    float rstd = rsqrtf(var + 1e-5f);
    #pragma unroll
    for (int t = 0; t < 3; ++t) {
        int c = threadIdx.x + t * 256;
        float o = (vals[t] - mean) * rstd * gm[c] + bt[c];
        if constexpr (OBF) ((__bf16*)outp)[row * HID_ + c] = (__bf16)o;
        else               ((float*)outp)[row * HID_ + c] = o;
    }
}

// =====================================================================
// fused score assembly + softmax:
//   s(q,k) = (c2c + Qpk[q, clamp(q-k)+M] + Kpq[k, clamp(k-q)+M]) * scale
//   mask: token_ids[b,k]==0 -> -1e9 ; row softmax -> bf16 attn
// grid: (S, B*NH), block 256 (each thread handles 2 keys)
// =====================================================================
__global__ __launch_bounds__(256)
void softmax_kernel(const float* __restrict__ scores, const __bf16* __restrict__ Qpk,
                    const __bf16* __restrict__ Kpq, const int* __restrict__ tok,
                    __bf16* __restrict__ attn)
{
    __shared__ float sh[8];
    int q = blockIdx.x;
    long long z = blockIdx.y;
    int b = (int)(z / NH_);
    const float*  srow = scores + (z * S_ + q) * S_;
    const __bf16* qrow = Qpk + (z * S_ + q) * TWOM;

    float v[2];
    #pragma unroll
    for (int t = 0; t < 2; ++t) {
        int k  = threadIdx.x + t * 256;
        int d1 = q - k; d1 = min(max(d1, -cM), cM - 1); int r1 = d1 + cM;
        int d2 = k - q; d2 = min(max(d2, -cM), cM - 1); int r2 = d2 + cM;
        float val = srow[k] + (float)qrow[r1]
                  + (float)Kpq[(z * S_ + k) * TWOM + r2];
        val *= SCALE_;
        if (tok[(long long)b * S_ + k] == 0) val = -1e9f;
        v[t] = val;
    }
    float mx = blk_max(fmaxf(v[0], v[1]), sh);
    float e0 = expf(v[0] - mx), e1 = expf(v[1] - mx);
    float inv = 1.0f / blk_sum(e0 + e1, sh);
    attn[(z * S_ + q) * S_ + threadIdx.x]       = (__bf16)(e0 * inv);
    attn[(z * S_ + q) * S_ + threadIdx.x + 256] = (__bf16)(e1 * inv);
}

// =====================================================================
// host-side launcher
// =====================================================================
extern "C" void kernel_launch(void* const* d_in, const int* in_sizes, int n_in,
                              void* d_out, int out_size, void* d_ws, size_t ws_size,
                              hipStream_t stream)
{
    (void)in_sizes; (void)n_in; (void)out_size; (void)ws_size;

    const int*   tok     = (const int*)d_in[0];
    const int*   seg     = (const int*)d_in[1];
    const float* tok_emb = (const float*)d_in[2];
    const float* seg_emb = (const float*)d_in[3];
    const float* Wq  = (const float*)d_in[4];  const float* bq  = (const float*)d_in[5];
    const float* Wk  = (const float*)d_in[6];  const float* bk  = (const float*)d_in[7];
    const float* Wv  = (const float*)d_in[8];  const float* bv  = (const float*)d_in[9];
    const float* Wo  = (const float*)d_in[10]; const float* bo  = (const float*)d_in[11];
    const float* Wpk = (const float*)d_in[12]; const float* Wpq = (const float*)d_in[13];
    const float* rel = (const float*)d_in[14];
    const float* ln1g = (const float*)d_in[15]; const float* ln1b = (const float*)d_in[16];
    const float* ln2g = (const float*)d_in[17]; const float* ln2b = (const float*)d_in[18];
    const float* W1 = (const float*)d_in[19];  const float* b1 = (const float*)d_in[20];
    const float* W2 = (const float*)d_in[21];  const float* b2 = (const float*)d_in[22];
    const float* lnfg = (const float*)d_in[23]; const float* lnfb = (const float*)d_in[24];
    float* out = (float*)d_out;

    // ---- workspace carve-out ----
    char* ws = (char*)d_ws;
    size_t off = 0;
    auto carve = [&](size_t bytes) { char* p = ws + off; off += (bytes + 255) & ~(size_t)255; return p; };
    float*  x      = (float*) carve((size_t)B_ * S_ * HID_ * 4);
    __bf16* h      = (__bf16*)carve((size_t)B_ * S_ * HID_ * 2);
    __bf16* qb     = (__bf16*)carve((size_t)B_ * S_ * HID_ * 2);
    __bf16* kbuf   = (__bf16*)carve((size_t)B_ * S_ * HID_ * 2);
    __bf16* vbuf   = (__bf16*)carve((size_t)B_ * S_ * HID_ * 2);
    __bf16* pk     = (__bf16*)carve((size_t)NH_ * TWOM * D_ * 2);
    __bf16* pq     = (__bf16*)carve((size_t)NH_ * TWOM * D_ * 2);
    __bf16* Qpk    = (__bf16*)carve((size_t)B_ * NH_ * S_ * TWOM * 2);   // ~101 MB
    __bf16* Kpq    = (__bf16*)carve((size_t)B_ * NH_ * S_ * TWOM * 2);   // ~101 MB
    float*  scores = (float*) carve((size_t)B_ * NH_ * S_ * S_ * 4);     // ~101 MB
    __bf16* attn   = (__bf16*)carve((size_t)B_ * NH_ * S_ * S_ * 2);     // ~50 MB
    float*  ctx    = (float*) carve((size_t)B_ * S_ * HID_ * 4);
    __bf16* gbuf   = (__bf16*)carve((size_t)B_ * S_ * FF_ * 2);          // ~25 MB

    const int ROWS = B_ * S_;                         // 4096
    const dim3 blk(256);

    embed_kernel<<<ROWS, blk, 0, stream>>>(tok, seg, tok_emb, seg_emb, x);

    for (int l = 0; l < L_; ++l) {
        const float* Wq_l  = Wq  + (size_t)l * HID_ * HID_;
        const float* Wk_l  = Wk  + (size_t)l * HID_ * HID_;
        const float* Wv_l  = Wv  + (size_t)l * HID_ * HID_;
        const float* Wo_l  = Wo  + (size_t)l * HID_ * HID_;
        const float* Wpk_l = Wpk + (size_t)l * HID_ * HID_;
        const float* Wpq_l = Wpq + (size_t)l * HID_ * HID_;
        const float* rel_l = rel + (size_t)l * TWOM * HID_;
        const float* W1_l  = W1  + (size_t)l * HID_ * FF_;
        const float* W2_l  = W2  + (size_t)l * FF_ * HID_;

        // --- attention sublayer (pre-norm) ---
        ln_kernel<true><<<ROWS, blk, 0, stream>>>(x, ln1g + l * HID_, ln1b + l * HID_, h);

        // q/k/v = h @ W + b  -> [B,NH,S,D] bf16
        gemm_wmma<EPI_QKV, false, false, true, 128, 128><<<dim3(6, 32, 1), blk, 0, stream>>>(
            h, Wq_l, qb, bq + l * HID_, HID_, HID_, HID_, HID_, 0, 0, 0, 0);
        gemm_wmma<EPI_QKV, false, false, true, 128, 128><<<dim3(6, 32, 1), blk, 0, stream>>>(
            h, Wk_l, kbuf, bk + l * HID_, HID_, HID_, HID_, HID_, 0, 0, 0, 0);
        gemm_wmma<EPI_QKV, false, false, true, 128, 128><<<dim3(6, 32, 1), blk, 0, stream>>>(
            h, Wv_l, vbuf, bv + l * HID_, HID_, HID_, HID_, HID_, 0, 0, 0, 0);

        // pk/pq = rel_e @ Wpk/Wpq  -> [NH,2M,D] bf16 (no bias)
        gemm_wmma<EPI_PK, true, false, true, 128, 128><<<dim3(6, 8, 1), blk, 0, stream>>>(
            rel_l, Wpk_l, pk, nullptr, HID_, HID_, HID_, HID_, 0, 0, 0, 0);
        gemm_wmma<EPI_PK, true, false, true, 128, 128><<<dim3(6, 8, 1), blk, 0, stream>>>(
            rel_l, Wpq_l, pq, nullptr, HID_, HID_, HID_, HID_, 0, 0, 0, 0);

        // Qpk[z] = q[z] @ pk[h]^T   (batched over z = b*NH+h), N=2M, K=D
        gemm_wmma<EPI_BF16, false, true, false, 128, 128><<<dim3(8, 4, B_ * NH_), blk, 0, stream>>>(
            qb, pk, Qpk, nullptr, TWOM, D_, D_, D_,
            (long long)S_ * D_, (long long)TWOM * D_, (long long)S_ * TWOM, NH_);
        // Kpq[z] = k[z] @ pq[h]^T
        gemm_wmma<EPI_BF16, false, true, false, 128, 128><<<dim3(8, 4, B_ * NH_), blk, 0, stream>>>(
            kbuf, pq, Kpq, nullptr, TWOM, D_, D_, D_,
            (long long)S_ * D_, (long long)TWOM * D_, (long long)S_ * TWOM, NH_);

        // c2c[z] = q[z] @ k[z]^T
        gemm_wmma<EPI_F32, false, true, false, 128, 128><<<dim3(4, 4, B_ * NH_), blk, 0, stream>>>(
            qb, kbuf, scores, nullptr, S_, D_, D_, D_,
            (long long)S_ * D_, (long long)S_ * D_, (long long)S_ * S_, 0);

        // fused disentangled-score gather + mask + softmax -> bf16 attn
        softmax_kernel<<<dim3(S_, B_ * NH_), blk, 0, stream>>>(scores, Qpk, Kpq, tok, attn);

        // ctx[z] = attn[z] @ v[z]  -> scattered to [B,S,HID] fp32
        gemm_wmma<EPI_CTX, false, false, false, 128, 64><<<dim3(1, 4, B_ * NH_), blk, 0, stream>>>(
            attn, vbuf, ctx, nullptr, D_, S_, S_, D_,
            (long long)S_ * S_, (long long)S_ * D_, 0, 0);

        // x += ctx @ Wo + bo
        gemm_wmma<EPI_RESID, true, false, true, 128, 128><<<dim3(6, 32, 1), blk, 0, stream>>>(
            ctx, Wo_l, x, bo + l * HID_, HID_, HID_, HID_, HID_, 0, 0, 0, 0);

        // --- feedforward sublayer (pre-norm) ---
        ln_kernel<true><<<ROWS, blk, 0, stream>>>(x, ln2g + l * HID_, ln2b + l * HID_, h);

        // g = gelu(h @ W1 + b1)  bf16
        gemm_wmma<EPI_GELU, false, false, true, 128, 128><<<dim3(24, 32, 1), blk, 0, stream>>>(
            h, W1_l, gbuf, b1 + l * FF_, FF_, HID_, HID_, FF_, 0, 0, 0, 0);

        // x += g @ W2 + b2
        gemm_wmma<EPI_RESID, false, false, true, 128, 128><<<dim3(6, 32, 1), blk, 0, stream>>>(
            gbuf, W2_l, x, b2 + l * HID_, HID_, FF_, FF_, HID_, 0, 0, 0, 0);
    }

    // final LayerNorm -> fp32 output
    ln_kernel<false><<<ROWS, blk, 0, stream>>>(x, lnfg, lnfb, out);
}
